// CNN_RNN_34514357190826
// MI455X (gfx1250) — matrix operand order repaired and checked by reading, not compiled, not run on hardware
//
#include <hip/hip_runtime.h>
#include <math.h>

// ---------------------------------------------------------------------------
// Problem constants (from the reference)
// ---------------------------------------------------------------------------
#define BATCH   8
#define KWIN    32             // (2176-128-1)/64 + 1
#define NIMG    (BATCH*KWIN)   // 256 CNN images
#define F1      257
#define TLEN    2176
#define NSHIFT  64

#define P1H 84                 // pool1 H  (conv1: 256+4-9+1=252, /3)
#define P1W 41                 // pool1 W  (conv1: 128+4-9+1=124, /3)
#define P2H 28
#define P2W 14

#define HID  3136              // 8*28*14
#define INP  6272              // 16*28*14
#define G3   (3*HID)           // 9408
#define NT_G (G3/16)           // 588 N-tiles (exact)

typedef __attribute__((ext_vector_type(16))) __bf16 v16bf;
typedef __attribute__((ext_vector_type(8)))  float  v8f;

union Frag16 { uint4  u[2];  v16bf v; };   // 16 bf16 = 32B
union F16x   { float4 f4[4]; float  f[16]; };

__device__ __forceinline__ unsigned short f2bf(float f) {
    unsigned int u = __float_as_uint(f);
    u += 0x7FFFu + ((u >> 16) & 1u);          // RNE
    return (unsigned short)(u >> 16);
}
__device__ __forceinline__ float sigmoidf(float x) {
    return 1.0f / (1.0f + __expf(-x));
}

// ---------------------------------------------------------------------------
// K0: window extraction + conv1(9x9,pad2) + leaky(0.01) + maxpool 3x3/3
// ---------------------------------------------------------------------------
__global__ void conv1_pool_kernel(const float* __restrict__ x,
                                  const float* __restrict__ w1,
                                  const float* __restrict__ b1,
                                  float* __restrict__ P1out) {
    __shared__ float wsh[81];
    const int img = blockIdx.x;          // 0..255
    const int oc  = blockIdx.y;          // 0..15
    for (int i = threadIdx.x; i < 81; i += blockDim.x)
        wsh[i] = w1[oc * 81 + i];
    __syncthreads();

    const int   b = img >> 5;
    const int   k = img & 31;
    const float bias = b1[oc];
    const float* xw = x + ((size_t)b * F1 + 1) * TLEN + (size_t)k * NSHIFT;

    for (int idx = threadIdx.x; idx < P1H * P1W; idx += blockDim.x) {
        const int ph = idx / P1W;
        const int pw = idx - ph * P1W;
        float mx = -1e30f;
        for (int dy = 0; dy < 3; ++dy) {
            const int oh = ph * 3 + dy;
            for (int dx = 0; dx < 3; ++dx) {
                const int ow = pw * 3 + dx;
                float s = bias;
                for (int kh = 0; kh < 9; ++kh) {
                    const int ih = oh + kh - 2;
                    if ((unsigned)ih < 256u) {
                        const float* xr = xw + (size_t)ih * TLEN;
                        const float* wr = wsh + kh * 9;
                        #pragma unroll
                        for (int kw = 0; kw < 9; ++kw) {
                            const int iw = ow + kw - 2;
                            if ((unsigned)iw < 128u) s += xr[iw] * wr[kw];
                        }
                    }
                }
                s = (s > 0.0f) ? s : 0.01f * s;
                mx = fmaxf(mx, s);
            }
        }
        P1out[(((size_t)img * 16 + oc) * P1H + ph) * P1W + pw] = mx;
    }
}

// ---------------------------------------------------------------------------
// K1: conv2(16ch,4x4,pad2) + leaky + maxpool 3x3/3 -> y_bf16 [NIMG][INP]
// ---------------------------------------------------------------------------
__global__ void conv2_pool_kernel(const float* __restrict__ P1in,
                                  const float* __restrict__ w2,
                                  const float* __restrict__ b2,
                                  unsigned short* __restrict__ ybf) {
    __shared__ float w2sh[16 * 16 * 16];
    const int img = blockIdx.x;
    for (int i = threadIdx.x; i < 4096; i += blockDim.x) w2sh[i] = w2[i];
    __syncthreads();

    const float* pimg = P1in + (size_t)img * 16 * P1H * P1W;

    for (int feat = threadIdx.x; feat < INP; feat += blockDim.x) {
        const int oc  = feat / (P2H * P2W);
        const int rem = feat - oc * (P2H * P2W);
        const int ph  = rem / P2W;
        const int pw  = rem - ph * P2W;
        float mx = -1e30f;
        for (int dy = 0; dy < 3; ++dy) {
            const int oh = ph * 3 + dy;
            for (int dx = 0; dx < 3; ++dx) {
                const int ow = pw * 3 + dx;
                float s = b2[oc];
                for (int ic = 0; ic < 16; ++ic) {
                    const float* base = pimg + (size_t)ic * P1H * P1W;
                    const float* wr   = w2sh + (oc * 16 + ic) * 16;
                    #pragma unroll
                    for (int kh = 0; kh < 4; ++kh) {
                        const int ih = oh + kh - 2;
                        if ((unsigned)ih < (unsigned)P1H) {
                            const float* br = base + (size_t)ih * P1W;
                            #pragma unroll
                            for (int kw = 0; kw < 4; ++kw) {
                                const int iw = ow + kw - 2;
                                if ((unsigned)iw < (unsigned)P1W)
                                    s += br[iw] * wr[kh * 4 + kw];
                            }
                        }
                    }
                }
                s = (s > 0.0f) ? s : 0.01f * s;
                mx = fmaxf(mx, s);
            }
        }
        ybf[(size_t)img * INP + feat] = f2bf(mx);
    }
}

// ---------------------------------------------------------------------------
// fp32 -> bf16 bulk conversion (vectorized, 8 elems/thread)
// ---------------------------------------------------------------------------
__global__ void cvt_bf16_kernel(const float* __restrict__ src,
                                unsigned short* __restrict__ dst, size_t n) {
    const size_t i = ((size_t)blockIdx.x * blockDim.x + threadIdx.x) * 8;
    if (i + 8 <= n) {
        const float4 a = *reinterpret_cast<const float4*>(src + i);
        const float4 b = *reinterpret_cast<const float4*>(src + i + 4);
        union { __bf16 h[8]; uint4 u; } o;
        o.h[0] = (__bf16)a.x; o.h[1] = (__bf16)a.y;
        o.h[2] = (__bf16)a.z; o.h[3] = (__bf16)a.w;
        o.h[4] = (__bf16)b.x; o.h[5] = (__bf16)b.y;
        o.h[6] = (__bf16)b.z; o.h[7] = (__bf16)b.w;
        *reinterpret_cast<uint4*>(dst + i) = o.u;
    } else {
        for (size_t j = i; j < n; ++j) dst[j] = f2bf(src[j]);
    }
}

// ---------------------------------------------------------------------------
// K_init: h (16 x HID fp32, rows 8..15 zero) + h_bf16
// ---------------------------------------------------------------------------
__global__ void init_h_kernel(const float* __restrict__ h0,
                              float* __restrict__ h,
                              unsigned short* __restrict__ hbf) {
    const int i = blockIdx.x * blockDim.x + threadIdx.x;
    if (i >= 16 * HID) return;
    const int row = i / HID;
    const float v = (row < BATCH) ? h0[i] : 0.0f;
    h[i]   = v;
    hbf[i] = f2bf(v);
}

// ---------------------------------------------------------------------------
// WMMA GEMM:  D[m][n] = sum_k A_bf16[m][k] * W[n][k] + bias[n]
//   One wave per N-tile; MT M-tiles per wave (B fragment reused MT times).
//   Double-buffered over K (K % 64 == 0 required): stage-1 loads issue
//   before stage-0 WMMAs, so matrix math hides L2 load latency.
//   WBF16: W already bf16 (pure b128 loads); else fp32 + in-register cvt.
//   Consistent K-permutation inside each 32-block -> all loads contiguous.
// ---------------------------------------------------------------------------
template<int MT, bool WBF16>
__global__ __launch_bounds__(32)
void gemm_wmma_kernel(const unsigned short* __restrict__ A, int lda,
                      const void* __restrict__ Wv,
                      const float* __restrict__ bias,
                      float* __restrict__ D, int ldd, int K) {
    const int  lane  = threadIdx.x;          // wave32
    const int  ntile = blockIdx.x;
    const int  mbase = blockIdx.y * MT;
    const int  l15   = lane & 15;
    const bool hi    = lane >= 16;
    const int  n     = ntile * 16 + l15;

    // A: low lanes {k0..+7, k0+16..+23}, high {k0+8..+15, k0+24..+31}
    const unsigned short* arow =
        A + (size_t)(mbase * 16 + l15) * lda + (hi ? 8 : 0);
    const size_t mstep = (size_t)16 * lda;   // elements between M-tiles

    const unsigned short* wrow_bf =
        WBF16 ? (const unsigned short*)Wv + (size_t)n * K + (hi ? 8 : 0) : nullptr;
    const float* wrow_f32 =
        WBF16 ? nullptr : (const float*)Wv + (size_t)n * K + (hi ? 8 : 0);

    v8f acc[MT];
    #pragma unroll
    for (int i = 0; i < MT; ++i) acc[i] = (v8f){};

    Frag16 af0[MT], af1[MT], bw0, bw1;

    auto load_stage = [&](Frag16 (&af)[MT], Frag16& bw, int k0) {
        if constexpr (WBF16) {
            // B: low lanes {k0..+7, k0+16..+23}, high {k0+8..+15, k0+24..+31}
            bw.u[0] = *reinterpret_cast<const uint4*>(wrow_bf + k0);
            bw.u[1] = *reinterpret_cast<const uint4*>(wrow_bf + k0 + 16);
        } else {
            F16x wf;
            wf.f4[0] = *reinterpret_cast<const float4*>(wrow_f32 + k0);
            wf.f4[1] = *reinterpret_cast<const float4*>(wrow_f32 + k0 + 4);
            wf.f4[2] = *reinterpret_cast<const float4*>(wrow_f32 + k0 + 16);
            wf.f4[3] = *reinterpret_cast<const float4*>(wrow_f32 + k0 + 20);
            union { __bf16 h[16]; Frag16 f; } cv;
            #pragma unroll
            for (int i = 0; i < 16; ++i) cv.h[i] = (__bf16)wf.f[i];
            bw = cv.f;
        }
        #pragma unroll
        for (int mt = 0; mt < MT; ++mt) {
            af[mt].u[0] = *reinterpret_cast<const uint4*>(arow + mt * mstep + k0);
            af[mt].u[1] = *reinterpret_cast<const uint4*>(arow + mt * mstep + k0 + 16);
        }
    };
    auto mma_stage = [&](Frag16 (&af)[MT], Frag16& bw) {
        #pragma unroll
        for (int mt = 0; mt < MT; ++mt)
            acc[mt] = __builtin_amdgcn_wmma_f32_16x16x32_bf16(
                          false, af[mt].v, false, bw.v, (short)0, acc[mt],
                          false, false);
    };

    // K % 64 == 0 for both GEMMs (6272, 3136); software pipeline depth 2.
    load_stage(af0, bw0, 0);
    for (int k0 = 0; k0 < K; k0 += 64) {
        load_stage(af1, bw1, k0 + 32);
        mma_stage(af0, bw0);
        if (k0 + 64 < K) load_stage(af0, bw0, k0 + 64);
        mma_stage(af1, bw1);
    }

    // C/D layout: vgpr r -> row r (low lanes) / 8+r (high lanes); col = lane&15
    const float bn = bias[n];
    #pragma unroll
    for (int mt = 0; mt < MT; ++mt) {
        float* dptr = D + (size_t)((mbase + mt) * 16 + (hi ? 8 : 0)) * ldd
                        + ntile * 16 + l15;
        #pragma unroll
        for (int r = 0; r < 8; ++r)
            dptr[(size_t)r * ldd] = acc[mt][r] + bn;
    }
}

// ---------------------------------------------------------------------------
// K4: GRU gate fusion for timestep t
// ---------------------------------------------------------------------------
__global__ void gru_fuse_kernel(const float* __restrict__ gi,
                                const float* __restrict__ gh,
                                float* __restrict__ h,
                                unsigned short* __restrict__ hbf,
                                int t) {
    const int i = blockIdx.x * blockDim.x + threadIdx.x;
    if (i >= BATCH * HID) return;
    const int b = i / HID;
    const int u = i - b * HID;
    const float* gir = gi + (size_t)(b * KWIN + t) * G3;
    const float* ghr = gh + (size_t)b * G3;

    const float r = sigmoidf(gir[u]          + ghr[u]);
    const float z = sigmoidf(gir[HID + u]    + ghr[HID + u]);
    const float n = tanhf  (gir[2 * HID + u] + r * ghr[2 * HID + u]);
    const float hv = h[(size_t)b * HID + u];
    const float hn = (1.0f - z) * n + z * hv;
    h[(size_t)b * HID + u]   = hn;
    hbf[(size_t)b * HID + u] = f2bf(hn);
}

// ---------------------------------------------------------------------------
// K5: out = h @ fc_w.T + fc_b -> (8,2)
// ---------------------------------------------------------------------------
__global__ void fc_kernel(const float* __restrict__ h,
                          const float* __restrict__ fcw,
                          const float* __restrict__ fcb,
                          float* __restrict__ out) {
    const int t = threadIdx.x;
    if (t >= BATCH * 2) return;
    const int b = t >> 1, o = t & 1;
    const float* hr = h + (size_t)b * HID;
    const float* wr = fcw + (size_t)o * HID;
    float s = fcb[o];
    for (int u = 0; u < HID; ++u) s += hr[u] * wr[u];
    out[b * 2 + o] = s;
}

// ---------------------------------------------------------------------------
// Host launcher
// ---------------------------------------------------------------------------
static inline size_t align_up(size_t v, size_t a) { return (v + a - 1) & ~(a - 1); }

extern "C" void kernel_launch(void* const* d_in, const int* in_sizes, int n_in,
                              void* d_out, int out_size, void* d_ws, size_t ws_size,
                              hipStream_t stream) {
    const float* x   = (const float*)d_in[0];
    const float* h0  = (const float*)d_in[1];
    const float* c1w = (const float*)d_in[2];
    const float* c1b = (const float*)d_in[3];
    const float* c2w = (const float*)d_in[4];
    const float* c2b = (const float*)d_in[5];
    const float* wih = (const float*)d_in[6];
    const float* whh = (const float*)d_in[7];
    const float* bih = (const float*)d_in[8];
    const float* bhh = (const float*)d_in[9];
    const float* fcw = (const float*)d_in[10];
    const float* fcb = (const float*)d_in[11];
    float* out = (float*)d_out;

    // ---- workspace layout: base (~70 MB) + optional bf16 weight copies ----
    char* p = (char*)d_ws;
    size_t off = 0;
    float* P1buf = (float*)(p + off);                 off = align_up(off + (size_t)NIMG*16*P1H*P1W*4, 256);
    unsigned short* ybf = (unsigned short*)(p + off); off = align_up(off + (size_t)NIMG*INP*2, 256);
    float* gi = (float*)(p + off);                    off = align_up(off + (size_t)NIMG*G3*4, 256);
    float* gh = (float*)(p + off);                    off = align_up(off + (size_t)16*G3*4, 256);
    float* h  = (float*)(p + off);                    off = align_up(off + (size_t)16*HID*4, 256);
    unsigned short* hbf = (unsigned short*)(p + off); off = align_up(off + (size_t)16*HID*2, 256);

    const size_t nWHH = (size_t)G3 * HID;             // 29.5M elems
    const size_t nWIH = (size_t)G3 * INP;             // 59.0M elems
    unsigned short* whh_bf = (unsigned short*)(p + off);
    const size_t off_whh_end = align_up(off + nWHH * 2, 256);
    unsigned short* wih_bf = (unsigned short*)(p + off_whh_end);
    const size_t off_wih_end = align_up(off_whh_end + nWIH * 2, 256);

    const bool use_whh_bf = (ws_size >= off_whh_end);   // +59 MB tier
    const bool use_wih_bf = (ws_size >= off_wih_end);   // +118 MB tier

    // ---- CNN front-end ----
    conv1_pool_kernel<<<dim3(NIMG, 16), 256, 0, stream>>>(x, c1w, c1b, P1buf);
    conv2_pool_kernel<<<NIMG, 256, 0, stream>>>(P1buf, c2w, c2b, ybf);

    // ---- hidden state init ----
    init_h_kernel<<<(16 * HID + 255) / 256, 256, 0, stream>>>(h0, h, hbf);

    // ---- one-time bf16 weight conversion (L2-resident reuse for w_hh) ----
    if (use_whh_bf)
        cvt_bf16_kernel<<<(unsigned)((nWHH / 8 + 255) / 256), 256, 0, stream>>>(
            whh, whh_bf, nWHH);
    if (use_wih_bf)
        cvt_bf16_kernel<<<(unsigned)((nWIH / 8 + 255) / 256), 256, 0, stream>>>(
            wih, wih_bf, nWIH);

    // ---- input-side GEMM hoisted out of the scan ----
    //   gi (256 x 9408) = y_bf (256 x 6272) @ w_ih^T + b_ih
    //   4 M-tiles per wave, double-buffered K; bf16 w_ih is L2-resident.
    if (use_wih_bf)
        gemm_wmma_kernel<4, true ><<<dim3(NT_G, 4), 32, 0, stream>>>(
            ybf, INP, wih_bf, bih, gi, G3, INP);
    else
        gemm_wmma_kernel<4, false><<<dim3(NT_G, 4), 32, 0, stream>>>(
            ybf, INP, wih, bih, gi, G3, INP);

    // ---- recurrence: 32 sequential (GEMM, fuse) pairs ----
    for (int t = 0; t < KWIN; ++t) {
        if (use_whh_bf)
            gemm_wmma_kernel<1, true ><<<dim3(NT_G, 1), 32, 0, stream>>>(
                hbf, HID, whh_bf, bhh, gh, G3, HID);
        else
            gemm_wmma_kernel<1, false><<<dim3(NT_G, 1), 32, 0, stream>>>(
                hbf, HID, whh, bhh, gh, G3, HID);
        gru_fuse_kernel<<<(BATCH * HID + 255) / 256, 256, 0, stream>>>(
            gi, gh, h, hbf, t);
    }

    // ---- final projection ----
    fc_kernel<<<1, 32, 0, stream>>>(h, fcw, fcb, out);
}
